// GCN_Fusion_surface_70669391888403
// MI455X (gfx1250) — matrix operand order repaired and checked by reading, not compiled, not run on hardware
//
#include <hip/hip_runtime.h>
#include <hip/hip_bf16.h>

typedef __attribute__((ext_vector_type(16))) _Float16 v16h;
typedef __attribute__((ext_vector_type(8)))  float    v8f;

#define DD 128
#define KSEL 101   // self + 100 neighbors; k=5/20/100 are prefixes after self

// ---------------------------------------------------------------------------
// Normalize direction matrices: dirs is (3 x 128) row-major; normalize columns.
__global__ __launch_bounds__(128) void norm_dirs_kernel(
    const float* __restrict__ dirs, float* __restrict__ out)
{
    int d = threadIdx.x;
    float a = dirs[d], b = dirs[128 + d], c = dirs[256 + d];
    float n = sqrtf(a * a + b * b + c * c);
    float inv = 1.0f / fmaxf(n, 1e-12f);
    out[d] = a * inv; out[128 + d] = b * inv; out[256 + d] = c * inv;
}

// ---------------------------------------------------------------------------
// KNN: one block per (batch, point). Distances in LDS, 101 iterative argmins.
__global__ __launch_bounds__(256) void knn_kernel(
    const float* __restrict__ v,   // B x N x 3
    int* __restrict__ idx,         // B x N x KSEL
    int N)
{
    __shared__ float dist[1024];
    __shared__ float rmin[256];
    __shared__ int   rmini[256];

    int b = blockIdx.x / N;
    int n = blockIdx.x - b * N;
    int t = threadIdx.x;
    const float* vb = v + (size_t)b * N * 3;
    float x0 = vb[n * 3 + 0], y0 = vb[n * 3 + 1], z0 = vb[n * 3 + 2];

    for (int j = t; j < N; j += 256) {
        float dx = vb[j * 3 + 0] - x0;
        float dy = vb[j * 3 + 1] - y0;
        float dz = vb[j * 3 + 2] - z0;
        dist[j] = dx * dx + dy * dy + dz * dz;
    }
    __syncthreads();

    int* outp = idx + ((size_t)b * N + n) * KSEL;
    for (int s = 0; s < KSEL; ++s) {
        float best = 3.0e38f; int besti = 0x7fffffff;
        for (int j = t; j < N; j += 256) {
            float d = dist[j];
            if (d < best || (d == best && j < besti)) { best = d; besti = j; }
        }
        rmin[t] = best; rmini[t] = besti;
        __syncthreads();
        for (int off = 128; off > 0; off >>= 1) {
            if (t < off) {
                float o = rmin[t + off]; int oi = rmini[t + off];
                if (o < rmin[t] || (o == rmin[t] && oi < rmini[t])) {
                    rmin[t] = o; rmini[t] = oi;
                }
            }
            __syncthreads();
        }
        if (t == 0) { outp[s] = rmini[0]; dist[rmini[0]] = 3.0e38f; }
        __syncthreads();
    }
}

// ---------------------------------------------------------------------------
// Surface conv: out[b,n,d] = max_j relu( dot(normalize(v[idx_j]-v[n]), dir_d) )
__global__ __launch_bounds__(128) void conv_surface_kernel(
    const int* __restrict__ idx, int koff, int k,
    const float* __restrict__ v,
    const float* __restrict__ dirsn,   // normalized 3 x 128
    float* __restrict__ out, int N)
{
    __shared__ float ndx[100], ndy[100], ndz[100];
    int b = blockIdx.x / N;
    int n = blockIdx.x - b * N;
    int t = threadIdx.x;
    const float* vb = v + (size_t)b * N * 3;
    const int* ip = idx + ((size_t)b * N + n) * KSEL + koff;
    float x0 = vb[n * 3 + 0], y0 = vb[n * 3 + 1], z0 = vb[n * 3 + 2];

    if (t < k) {
        int j = ip[t];
        float dx = vb[j * 3 + 0] - x0;
        float dy = vb[j * 3 + 1] - y0;
        float dz = vb[j * 3 + 2] - z0;
        float inv = rsqrtf(fmaxf(dx * dx + dy * dy + dz * dz, 1e-24f));
        ndx[t] = dx * inv; ndy[t] = dy * inv; ndz[t] = dz * inv;
    }
    __syncthreads();

    float d0 = dirsn[t], d1 = dirsn[128 + t], d2 = dirsn[256 + t];
    float m = 0.0f;  // relu'd values are >= 0
    for (int j = 0; j < k; ++j) {
        float th = fmaxf(ndx[j] * d0 + ndy[j] * d1 + ndz[j] * d2, 0.0f);
        m = fmaxf(m, th);
    }
    out[((size_t)b * N + n) * DD + t] = m;
}

// ---------------------------------------------------------------------------
// Conv layer tail: out = fc + max_j( theta_j * fs[idx_j] )
// fo is (B*N) x 256 row-major: cols 0..127 = fc, 128..255 = fs
__global__ __launch_bounds__(128) void conv_layer_rest_kernel(
    const int* __restrict__ idx, int koff, int k,
    const float* __restrict__ v,
    const float* __restrict__ dirsn,
    const float* __restrict__ fo,
    float* __restrict__ out, int N)
{
    __shared__ float ndx[100], ndy[100], ndz[100];
    __shared__ int   nj[100];
    int b = blockIdx.x / N;
    int n = blockIdx.x - b * N;
    int t = threadIdx.x;
    const float* vb = v + (size_t)b * N * 3;
    const int* ip = idx + ((size_t)b * N + n) * KSEL + koff;
    float x0 = vb[n * 3 + 0], y0 = vb[n * 3 + 1], z0 = vb[n * 3 + 2];

    if (t < k) {
        int j = ip[t];
        nj[t] = j;
        float dx = vb[j * 3 + 0] - x0;
        float dy = vb[j * 3 + 1] - y0;
        float dz = vb[j * 3 + 2] - z0;
        float inv = rsqrtf(fmaxf(dx * dx + dy * dy + dz * dz, 1e-24f));
        ndx[t] = dx * inv; ndy[t] = dy * inv; ndz[t] = dz * inv;
    }
    __syncthreads();

    float d0 = dirsn[t], d1 = dirsn[128 + t], d2 = dirsn[256 + t];
    float m = -3.0e38f;
    for (int j = 0; j < k; ++j) {
        float th = fmaxf(ndx[j] * d0 + ndy[j] * d1 + ndz[j] * d2, 0.0f);
        float fs = fo[((size_t)b * N + nj[j]) * 256 + 128 + t];
        m = fmaxf(m, th * fs);
    }
    size_t row = (size_t)b * N + n;
    out[row * DD + t] = fo[row * 256 + t] + m;
}

// ---------------------------------------------------------------------------
// BatchNorm stats: one block per channel d (of 128), reduce over `rows`.
__global__ __launch_bounds__(256) void bn_stats_kernel(
    const float* __restrict__ x, int rows,
    float* __restrict__ meanv, float* __restrict__ rstd)
{
    __shared__ float ssum[256], ssq[256];
    int d = blockIdx.x, t = threadIdx.x;
    float s = 0.0f, q = 0.0f;
    for (int r = t; r < rows; r += 256) {
        float val = x[(size_t)r * DD + d];
        s += val; q += val * val;
    }
    ssum[t] = s; ssq[t] = q;
    __syncthreads();
    for (int off = 128; off > 0; off >>= 1) {
        if (t < off) { ssum[t] += ssum[t + off]; ssq[t] += ssq[t + off]; }
        __syncthreads();
    }
    if (t == 0) {
        float m = ssum[0] / (float)rows;
        float var = ssq[0] / (float)rows - m * m;
        meanv[d] = m;
        rstd[d]  = rsqrtf(var + 1e-5f);
    }
}

// bn+relu apply; writes into out at column offset (enables direct concat fill)
__global__ __launch_bounds__(256) void bn_apply_kernel(
    const float* __restrict__ x,
    const float* __restrict__ meanv, const float* __restrict__ rstd,
    const float* __restrict__ g, const float* __restrict__ be,
    float* __restrict__ out, int rows, int out_ld, int out_coff)
{
    int i = blockIdx.x * 256 + threadIdx.x;
    if (i >= rows * DD) return;
    int r = i >> 7, d = i & 127;
    float val = (x[i] - meanv[d]) * rstd[d] * g[d] + be[d];
    out[(size_t)r * out_ld + out_coff + d] = fmaxf(val, 0.0f);
}

// ---------------------------------------------------------------------------
// Pack f32 A (MxK row-major) into WMMA-A fragment order (f16):
// Ap[((tm*(K/32)+kk)*32 + lane)*16 + i] = A[tm*16 + (lane&15)][k(lane,i)]
// so each lane's fragment is one contiguous 32-byte chunk.
__global__ __launch_bounds__(256) void pack_a_f16_kernel(
    const float* __restrict__ A, _Float16* __restrict__ Ap, int M, int K)
{
    int i = blockIdx.x * 256 + threadIdx.x;
    if (i >= M * K) return;
    int ii   = i & 15;          // element within fragment
    int lane = (i >> 4) & 31;
    int frag = i >> 9;          // (tm, kk) pair
    int KS   = K >> 5;
    int tm   = frag / KS;
    int kk   = frag - tm * KS;
    int r    = ii >> 1;
    int k    = (kk << 5) + ((r >> 2) << 4) + ((lane >> 4) << 3) + ((r & 3) << 1) + (ii & 1);
    int row  = (tm << 4) + (lane & 15);
    Ap[i] = (_Float16)A[(size_t)row * K + k];
}

// Pack f32 B (KxN row-major) into WMMA-B fragment order (f16):
// Bp[((tn*(K/32)+kk)*32 + lane)*16 + i] = B[krow(lane,i)][tn*16 + (lane&15)]
__global__ __launch_bounds__(256) void pack_b_f16_kernel(
    const float* __restrict__ Bsrc, _Float16* __restrict__ Bp, int N, int K)
{
    int i = blockIdx.x * 256 + threadIdx.x;
    if (i >= N * K) return;
    int ii   = i & 15;
    int lane = (i >> 4) & 31;
    int frag = i >> 9;          // (tn, kk) pair
    int KS   = K >> 5;
    int tn   = frag / KS;
    int kk   = frag - tn * KS;
    int r    = ii >> 1;
    int krow = (kk << 5) + ((r >> 2) << 4) + ((lane >> 4) << 3) + ((r & 3) << 1) + (ii & 1);
    int col  = (tn << 4) + (lane & 15);
    Bp[i] = (_Float16)Bsrc[(size_t)krow * N + col];
}

// ---------------------------------------------------------------------------
// WMMA GEMM on pre-packed fragments: C(MxN,f32) = A*B + bias.
// One wave computes a 32x64 tile with software pipelining: the 12 b128 loads
// of k-step kk+1 are issued before the 8 WMMAs of step kk, so VMEM latency
// overlaps the matrix pipe within a single wave.
__global__ __launch_bounds__(32) void gemm_f16_wmma_kernel(
    const _Float16* __restrict__ Ap,   // packed (M/16, K/32, 32 lanes, 16)
    const _Float16* __restrict__ Bp,   // packed (N/16, K/32, 32 lanes, 16)
    const float* __restrict__ bias,
    float* __restrict__ C,
    int M, int N, int K, int relu)
{
    int KS   = K >> 5;
    int ntn  = N >> 6;                    // 64-wide N tiles
    int tm2  = blockIdx.x / ntn;          // group of 2 16-row m-tiles
    int tn4  = blockIdx.x - tm2 * ntn;    // group of 4 16-col n-tiles
    int lane = threadIdx.x & 31;

    v8f acc[2][4] = {{v8f{}, v8f{}, v8f{}, v8f{}},
                     {v8f{}, v8f{}, v8f{}, v8f{}}};

    const _Float16* apb0 = Ap + ((size_t)(tm2 * 2 + 0) * KS) * 512 + lane * 16;
    const _Float16* apb1 = Ap + ((size_t)(tm2 * 2 + 1) * KS) * 512 + lane * 16;
    const _Float16* bpb  = Bp + ((size_t)(tn4 * 4) * KS) * 512 + lane * 16;

    // prologue: load k-step 0 fragments
    v16h a0 = *(const v16h*)(apb0);
    v16h a1 = *(const v16h*)(apb1);
    v16h b0 = *(const v16h*)(bpb + ((size_t)0 * KS) * 512);
    v16h b1 = *(const v16h*)(bpb + ((size_t)1 * KS) * 512);
    v16h b2 = *(const v16h*)(bpb + ((size_t)2 * KS) * 512);
    v16h b3 = *(const v16h*)(bpb + ((size_t)3 * KS) * 512);

    for (int kk = 0; kk < KS; ++kk) {
        // prefetch k-step kk+1 (clamped: last step re-loads itself, harmless)
        int nk = (kk + 1 < KS) ? (kk + 1) : kk;
        v16h na0 = *(const v16h*)(apb0 + (size_t)nk * 512);
        v16h na1 = *(const v16h*)(apb1 + (size_t)nk * 512);
        v16h nb0 = *(const v16h*)(bpb + ((size_t)0 * KS + nk) * 512);
        v16h nb1 = *(const v16h*)(bpb + ((size_t)1 * KS + nk) * 512);
        v16h nb2 = *(const v16h*)(bpb + ((size_t)2 * KS + nk) * 512);
        v16h nb3 = *(const v16h*)(bpb + ((size_t)3 * KS + nk) * 512);

        acc[0][0] = __builtin_amdgcn_wmma_f32_16x16x32_f16(false, a0, false, b0, (short)0, acc[0][0], false, false);
        acc[0][1] = __builtin_amdgcn_wmma_f32_16x16x32_f16(false, a0, false, b1, (short)0, acc[0][1], false, false);
        acc[0][2] = __builtin_amdgcn_wmma_f32_16x16x32_f16(false, a0, false, b2, (short)0, acc[0][2], false, false);
        acc[0][3] = __builtin_amdgcn_wmma_f32_16x16x32_f16(false, a0, false, b3, (short)0, acc[0][3], false, false);
        acc[1][0] = __builtin_amdgcn_wmma_f32_16x16x32_f16(false, a1, false, b0, (short)0, acc[1][0], false, false);
        acc[1][1] = __builtin_amdgcn_wmma_f32_16x16x32_f16(false, a1, false, b1, (short)0, acc[1][1], false, false);
        acc[1][2] = __builtin_amdgcn_wmma_f32_16x16x32_f16(false, a1, false, b2, (short)0, acc[1][2], false, false);
        acc[1][3] = __builtin_amdgcn_wmma_f32_16x16x32_f16(false, a1, false, b3, (short)0, acc[1][3], false, false);

        a0 = na0; a1 = na1; b0 = nb0; b1 = nb1; b2 = nb2; b3 = nb3;
    }

    // D layout: VGPR r -> row (lane>>4)*8 + r, col lane&15
    int half = lane >> 4, l15 = lane & 15;
#pragma unroll
    for (int m = 0; m < 2; ++m) {
#pragma unroll
        for (int t = 0; t < 4; ++t) {
            int coln = (tn4 * 4 + t) * 16 + l15;
            float bv = bias ? bias[coln] : 0.0f;
#pragma unroll
            for (int r = 0; r < 8; ++r) {
                int row = (tm2 * 2 + m) * 16 + (half << 3) + r;
                float v = acc[m][t][r] + bv;
                if (relu) v = fmaxf(v, 0.0f);
                C[(size_t)row * N + coln] = v;
            }
        }
    }
}

// ---------------------------------------------------------------------------
extern "C" void kernel_launch(void* const* d_in, const int* in_sizes, int n_in,
                              void* d_out, int out_size, void* d_ws, size_t ws_size,
                              hipStream_t stream)
{
    (void)in_sizes; (void)n_in; (void)out_size; (void)ws_size;
    const int B = 4, N = 1024, R = B * N;

    const float* verts  = (const float*)d_in[0];
    const float* dirs_l = (const float*)d_in[1];
    const float* dirs_m0= (const float*)d_in[2];
    const float* W_m1   = (const float*)d_in[3];
    const float* b_m1   = (const float*)d_in[4];
    const float* dirs_m1= (const float*)d_in[5];
    const float* dirs_g0= (const float*)d_in[6];
    const float* W_g1   = (const float*)d_in[7];
    const float* b_g1   = (const float*)d_in[8];
    const float* dirs_g1= (const float*)d_in[9];
    const float* W_g2   = (const float*)d_in[10];
    const float* b_g2   = (const float*)d_in[11];
    const float* dirs_g2= (const float*)d_in[12];
    const float* g_l  = (const float*)d_in[13]; const float* be_l  = (const float*)d_in[14];
    const float* g_m0 = (const float*)d_in[15]; const float* be_m0 = (const float*)d_in[16];
    const float* g_m1 = (const float*)d_in[17]; const float* be_m1 = (const float*)d_in[18];
    const float* g_g0 = (const float*)d_in[19]; const float* be_g0 = (const float*)d_in[20];
    const float* g_g1 = (const float*)d_in[21]; const float* be_g1 = (const float*)d_in[22];
    const float* g_g2 = (const float*)d_in[23]; const float* be_g2 = (const float*)d_in[24];
    const float* W_down = (const float*)d_in[25];
    const float* b_down = (const float*)d_in[26];
    float* out = (float*)d_out;

    // bump allocator over d_ws
    char* ws = (char*)d_ws;
    size_t off = 0;
    auto alloc = [&](size_t bytes) -> void* {
        void* p = ws + off;
        off = (off + bytes + 255) & ~(size_t)255;
        return p;
    };
    int*       idx   = (int*)      alloc((size_t)R * KSEL * 4);
    float*     dirsn = (float*)    alloc(6 * 384 * 4);
    float*     meanv = (float*)    alloc(128 * 4);
    float*     rstdv = (float*)    alloc(128 * 4);
    float*     t0    = (float*)    alloc((size_t)R * 128 * 4);
    float*     f0    = (float*)    alloc((size_t)R * 128 * 4);
    float*     t1    = (float*)    alloc((size_t)R * 128 * 4);
    float*     fo    = (float*)    alloc((size_t)R * 256 * 4);
    float*     X     = (float*)    alloc((size_t)R * 384 * 4);
    _Float16*  Ap    = (_Float16*) alloc((size_t)R * 384 * 2);   // packed A fragments
    _Float16*  Bp    = (_Float16*) alloc((size_t)384 * 256 * 2); // packed B fragments

    const int eltGrid  = (R * 128 + 255) / 256;
    const int packA128 = (R * 128 + 255) / 256;
    const int packA384 = (R * 384 + 255) / 256;
    const int packB128 = (128 * 256 + 255) / 256;
    const int packB384 = (384 * 256 + 255) / 256;
    const int gemmGrid = (R / 32) * (256 / 64);   // 32x64 tiles -> 512 waves

    // normalized direction banks: l,m0,m1,g0,g1,g2
    norm_dirs_kernel<<<1, 128, 0, stream>>>(dirs_l,  dirsn + 0 * 384);
    norm_dirs_kernel<<<1, 128, 0, stream>>>(dirs_m0, dirsn + 1 * 384);
    norm_dirs_kernel<<<1, 128, 0, stream>>>(dirs_m1, dirsn + 2 * 384);
    norm_dirs_kernel<<<1, 128, 0, stream>>>(dirs_g0, dirsn + 3 * 384);
    norm_dirs_kernel<<<1, 128, 0, stream>>>(dirs_g1, dirsn + 4 * 384);
    norm_dirs_kernel<<<1, 128, 0, stream>>>(dirs_g2, dirsn + 5 * 384);

    knn_kernel<<<R, 256, 0, stream>>>(verts, idx, N);

    // ---- local branch (k=5): surface -> bn_relu -> X[:, 0:128]
    conv_surface_kernel<<<R, 128, 0, stream>>>(idx, 1, 5, verts, dirsn + 0 * 384, t0, N);
    bn_stats_kernel<<<128, 256, 0, stream>>>(t0, R, meanv, rstdv);
    bn_apply_kernel<<<eltGrid, 256, 0, stream>>>(t0, meanv, rstdv, g_l, be_l, X, R, 384, 0);

    // ---- medium branch (k=20): surface -> bn -> linear -> conv tail -> bn -> X[:,128:256]
    conv_surface_kernel<<<R, 128, 0, stream>>>(idx, 1, 20, verts, dirsn + 1 * 384, t0, N);
    bn_stats_kernel<<<128, 256, 0, stream>>>(t0, R, meanv, rstdv);
    bn_apply_kernel<<<eltGrid, 256, 0, stream>>>(t0, meanv, rstdv, g_m0, be_m0, f0, R, 128, 0);
    pack_a_f16_kernel<<<packA128, 256, 0, stream>>>(f0, Ap, R, 128);
    pack_b_f16_kernel<<<packB128, 256, 0, stream>>>(W_m1, Bp, 256, 128);
    gemm_f16_wmma_kernel<<<gemmGrid, 32, 0, stream>>>(Ap, Bp, b_m1, fo, R, 256, 128, 0);
    conv_layer_rest_kernel<<<R, 128, 0, stream>>>(idx, 1, 20, verts, dirsn + 2 * 384, fo, t1, N);
    bn_stats_kernel<<<128, 256, 0, stream>>>(t1, R, meanv, rstdv);
    bn_apply_kernel<<<eltGrid, 256, 0, stream>>>(t1, meanv, rstdv, g_m1, be_m1, X, R, 384, 128);

    // ---- global branch (k=100): surface -> bn -> 2x(linear -> conv tail -> bn) -> X[:,256:384]
    conv_surface_kernel<<<R, 128, 0, stream>>>(idx, 1, 100, verts, dirsn + 3 * 384, t0, N);
    bn_stats_kernel<<<128, 256, 0, stream>>>(t0, R, meanv, rstdv);
    bn_apply_kernel<<<eltGrid, 256, 0, stream>>>(t0, meanv, rstdv, g_g0, be_g0, f0, R, 128, 0);

    pack_a_f16_kernel<<<packA128, 256, 0, stream>>>(f0, Ap, R, 128);
    pack_b_f16_kernel<<<packB128, 256, 0, stream>>>(W_g1, Bp, 256, 128);
    gemm_f16_wmma_kernel<<<gemmGrid, 32, 0, stream>>>(Ap, Bp, b_g1, fo, R, 256, 128, 0);
    conv_layer_rest_kernel<<<R, 128, 0, stream>>>(idx, 1, 100, verts, dirsn + 4 * 384, fo, t1, N);
    bn_stats_kernel<<<128, 256, 0, stream>>>(t1, R, meanv, rstdv);
    bn_apply_kernel<<<eltGrid, 256, 0, stream>>>(t1, meanv, rstdv, g_g1, be_g1, f0, R, 128, 0);

    pack_a_f16_kernel<<<packA128, 256, 0, stream>>>(f0, Ap, R, 128);
    pack_b_f16_kernel<<<packB128, 256, 0, stream>>>(W_g2, Bp, 256, 128);
    gemm_f16_wmma_kernel<<<gemmGrid, 32, 0, stream>>>(Ap, Bp, b_g2, fo, R, 256, 128, 0);
    conv_layer_rest_kernel<<<R, 128, 0, stream>>>(idx, 1, 100, verts, dirsn + 5 * 384, fo, t1, N);
    bn_stats_kernel<<<128, 256, 0, stream>>>(t1, R, meanv, rstdv);
    bn_apply_kernel<<<eltGrid, 256, 0, stream>>>(t1, meanv, rstdv, g_g2, be_g2, X, R, 384, 256);

    // ---- head: relu(X(4096x384) @ W_down(384x256) + b_down) -> d_out
    pack_a_f16_kernel<<<packA384, 256, 0, stream>>>(X, Ap, R, 384);
    pack_b_f16_kernel<<<packB384, 256, 0, stream>>>(W_down, Bp, 256, 384);
    gemm_f16_wmma_kernel<<<gemmGrid, 32, 0, stream>>>(Ap, Bp, b_down, out, R, 256, 384, 1);
}
